// Encoder_2645699854337
// MI455X (gfx1250) — compile-verified, matrix operand order checked by reading, not run on hardware
//
#include <hip/hip_runtime.h>

// ---------------------------------------------------------------------------
// VGAE encoder (2x GCN) for MI455X / gfx1250.
//   * GEMMs: v_wmma_f32_16x16x32_bf16, fp32 accumulate.
//       - A tiles staged in LDS as bf16 (converted once during staging);
//         each lane's 16x32 A fragment = two 16B ds_load_b128 (conflict-free,
//         row stride 528B).
//       - B (weights) pre-packed once into fragment-major order so each
//         lane's B fragment is ONE aligned 32B vector load (2x b128).
//   * Edge aggregation: float4 gather + global_atomic_add_f32 scatter
//     (1.7M edges; ~3.5GB of L2-resident traffic is the roofline limiter).
// ---------------------------------------------------------------------------

typedef __attribute__((ext_vector_type(16))) __bf16 v16bf;
typedef __attribute__((ext_vector_type(8)))  float  v8f;

#define LRELU_SLOPE 0.01f

union FragCast { v16bf v; uint4 q[2]; };

// ------------------------------ utility kernels ----------------------------

__global__ void fill_f32(float* __restrict__ p, float v, long long n) {
    long long i = (long long)blockIdx.x * blockDim.x + threadIdx.x;
    if (i < n) p[i] = v;
}

__global__ void degree_k(const long long* __restrict__ dstI,
                         float* __restrict__ deg, int E) {
    int e = blockIdx.x * blockDim.x + threadIdx.x;
    if (e < E) atomicAdd(&deg[(int)dstI[e]], 1.0f);
}

__global__ void rsqrt_k(float* __restrict__ deg, int n) {
    int i = blockIdx.x * blockDim.x + threadIdx.x;
    if (i < n) deg[i] = rsqrtf(deg[i]);   // deg >= 1 (self-loop pre-counted)
}

__global__ void norm_k(const long long* __restrict__ srcI,
                       const long long* __restrict__ dstI,
                       const float* __restrict__ dinv,
                       float* __restrict__ nrm, int E, int N) {
    int e = blockIdx.x * blockDim.x + threadIdx.x;
    if (e < E) {
        nrm[e] = dinv[(int)srcI[e]] * dinv[(int)dstI[e]];
    } else if (e < E + N) {
        float d = dinv[e - E];
        nrm[e] = d * d;                   // self-loop
    }
}

// Pack W[K x N] fp32 into WMMA B-fragment-major bf16:
//   Wpk[((nt*Ksteps + ks)*32 + lane)*16 + e]
// where (lane, e) follow the ISA 7.12.2 16-bit B layout for one 32x16 tile.
__global__ void pack_w_k(const float* __restrict__ W, __bf16* __restrict__ Wpk,
                         int K, int N) {
    int i = blockIdx.x * blockDim.x + threadIdx.x;
    if (i >= K * N) return;
    const int ksteps = K >> 5;
    int e    = i & 15;
    int l    = (i >> 4) & 31;
    int rest = i >> 9;
    int ks   = rest % ksteps;
    int nt   = rest / ksteps;
    int j = e >> 1, half = e & 1;
    int kg = l >> 4;
    int k   = (ks << 5) + ((j >> 2) << 4) + (kg << 3) + ((j & 3) << 1) + half;
    int col = (nt << 4) + (l & 15);
    Wpk[i] = (__bf16)W[k * N + col];
}

// -------- GEMM1: Hpre[M,128] = X[M,256] @ W1 (bf16 WMMA, fp32 acc) ---------
// Block: 256 threads = 8 waves; wave w owns N-tile w (N=128 -> 8 tiles).
// A tile converted to bf16 at staging time; LDS row stride 264 bf16 (528B,
// 16B-aligned, rows on distinct banks).

#define LDA1 264   // 256 + 8 pad (bf16 units)

__global__ __launch_bounds__(256) void gemm_x_w1(
        const float* __restrict__ X, const __bf16* __restrict__ Wpk,
        float* __restrict__ Out, int M) {
    __shared__ __bf16 AsB[16 * LDA1];
    const int m0 = blockIdx.x << 4;
    const int t  = threadIdx.x;

    #pragma unroll
    for (int it = 0; it < 4; ++it) {          // 16x256 = 4096 elems, 16/thread
        int idx = it * 1024 + t * 4;
        int r = idx >> 8, cc = idx & 255;
        int rr = m0 + r; if (rr >= M) rr = M - 1;      // clamp for tail tile
        float4 v4 = *(const float4*)(X + (long long)rr * 256 + cc);
        __bf16* d = &AsB[r * LDA1 + cc];               // 8B-aligned run of 4
        d[0] = (__bf16)v4.x; d[1] = (__bf16)v4.y;
        d[2] = (__bf16)v4.z; d[3] = (__bf16)v4.w;
    }
    __syncthreads();

    const int wv = t >> 5;        // N-tile id (0..7)
    const int l  = t & 31;
    const int hr = l & 15;        // A row / B,C column within tile
    const int kg = l >> 4;        // K half-group select (ISA 7.12.2)
    const __bf16* arow = &AsB[hr * LDA1 + (kg << 3)];
    v8f acc = {};

    #pragma unroll
    for (int ks = 0; ks < 8; ++ks) {
        // A fragment = two contiguous 8-elem K runs -> 2x ds_load_b128
        FragCast fa;
        fa.q[0] = *(const uint4*)(arow + (ks << 5));
        fa.q[1] = *(const uint4*)(arow + (ks << 5) + 16);
        // B fragment = one aligned 32B chunk -> 2x global_load_b128
        v16bf vb = *(const v16bf*)(Wpk + (((wv << 3) + ks) << 5 | l) * 16);
        acc = __builtin_amdgcn_wmma_f32_16x16x32_bf16(
                  false, fa.v, false, vb, (short)0, acc, false, false);
    }

    const int col = (wv << 4) + hr;
    #pragma unroll
    for (int vv = 0; vv < 8; ++vv) {
        int mrow = m0 + (kg << 3) + vv;                 // C layout: M = 8*kg+v
        if (mrow < M) Out[(long long)mrow * 128 + col] = acc[vv];
    }
}

// ------ Fused GEMM2/3: Mu[M,64] = H@Wmu, Lv[M,64] = H@Wlv (shared A) -------
// Waves 0-3 -> mu N-tiles, waves 4-7 -> logvar N-tiles. K = 128.

#define LDA2 136   // 128 + 8 pad (bf16 units)

__global__ __launch_bounds__(256) void gemm_h_w2(
        const float* __restrict__ H, const __bf16* __restrict__ WmuPk,
        const __bf16* __restrict__ WlvPk, float* __restrict__ OutMu,
        float* __restrict__ OutLv, int M) {
    __shared__ __bf16 AsB[16 * LDA2];
    const int m0 = blockIdx.x << 4;
    const int t  = threadIdx.x;

    #pragma unroll
    for (int it = 0; it < 2; ++it) {          // 16x128 = 2048 elems, 8/thread
        int idx = it * 1024 + t * 4;
        int r = idx >> 7, cc = idx & 127;
        int rr = m0 + r; if (rr >= M) rr = M - 1;
        float4 v4 = *(const float4*)(H + (long long)rr * 128 + cc);
        __bf16* d = &AsB[r * LDA2 + cc];
        d[0] = (__bf16)v4.x; d[1] = (__bf16)v4.y;
        d[2] = (__bf16)v4.z; d[3] = (__bf16)v4.w;
    }
    __syncthreads();

    const int wv = t >> 5;
    const __bf16* Wpk = (wv < 4) ? WmuPk : WlvPk;
    float*        Out = (wv < 4) ? OutMu : OutLv;
    const int nt = wv & 3;        // N-tile within the 64-wide output
    const int l  = t & 31;
    const int hr = l & 15;
    const int kg = l >> 4;
    const __bf16* arow = &AsB[hr * LDA2 + (kg << 3)];
    v8f acc = {};

    #pragma unroll
    for (int ks = 0; ks < 4; ++ks) {
        FragCast fa;
        fa.q[0] = *(const uint4*)(arow + (ks << 5));
        fa.q[1] = *(const uint4*)(arow + (ks << 5) + 16);
        v16bf vb = *(const v16bf*)(Wpk + (((nt << 2) + ks) << 5 | l) * 16);
        acc = __builtin_amdgcn_wmma_f32_16x16x32_bf16(
                  false, fa.v, false, vb, (short)0, acc, false, false);
    }

    const int col = (nt << 4) + hr;
    #pragma unroll
    for (int vv = 0; vv < 8; ++vv) {
        int mrow = m0 + (kg << 3) + vv;
        if (mrow < M) Out[(long long)mrow * 64 + col] = acc[vv];
    }
}

// --------- edge gather/scale/scatter: out[dst] += feat[src] * norm ---------
// One (edge, float4-chunk) pair per thread; F = 4 << c4shift.

__global__ void scatter_add(const float* __restrict__ feat,
                            const long long* __restrict__ srcI,
                            const long long* __restrict__ dstI,
                            const float* __restrict__ nrm,
                            float* __restrict__ out,
                            int E, int N, int c4shift) {
    long long t = (long long)blockIdx.x * blockDim.x + threadIdx.x;
    long long total = (long long)(E + N) << c4shift;
    if (t >= total) return;
    int e = (int)(t >> c4shift);
    int c = (int)(t & ((1 << c4shift) - 1));
    int s, d;
    if (e < E) { s = (int)srcI[e]; d = (int)dstI[e]; }
    else       { s = e - E; d = s; }                    // self-loop
    float w = nrm[e];
    const float4 mv = ((const float4*)(feat + ((long long)s << (c4shift + 2))))[c];
    float* o = out + ((long long)d << (c4shift + 2)) + (c << 2);
    atomicAdd(o + 0, mv.x * w);
    atomicAdd(o + 1, mv.y * w);
    atomicAdd(o + 2, mv.z * w);
    atomicAdd(o + 3, mv.w * w);
}

// ------------------------------ epilogues ----------------------------------

__global__ void finalize_h(float* __restrict__ h, const float* __restrict__ b,
                           long long n, int F) {
    long long i = (long long)blockIdx.x * blockDim.x + threadIdx.x;
    if (i < n) {
        float v = h[i] + b[(int)(i & (F - 1))];
        h[i] = (v >= 0.0f) ? v : LRELU_SLOPE * v;
    }
}

__global__ void finalize_out(float* __restrict__ out,
                             const float* __restrict__ bmu,
                             const float* __restrict__ blv,
                             long long M, int F) {
    long long i = (long long)blockIdx.x * blockDim.x + threadIdx.x;
    long long half = M * F;
    if (i < 2 * half) {
        const float* b = (i < half) ? bmu : blv;
        out[i] += b[(int)(i & (F - 1))];
    }
}

// ------------------------------ launcher -----------------------------------

extern "C" void kernel_launch(void* const* d_in, const int* in_sizes, int n_in,
                              void* d_out, int out_size, void* d_ws,
                              size_t ws_size, hipStream_t stream) {
    (void)n_in; (void)out_size; (void)ws_size;

    const float*     x    = (const float*)d_in[0];
    const long long* ei   = (const long long*)d_in[1];
    const float*     W1   = (const float*)d_in[2];
    const float*     b1   = (const float*)d_in[3];
    const float*     Wmu  = (const float*)d_in[4];
    const float*     bmu  = (const float*)d_in[5];
    const float*     Wlv  = (const float*)d_in[6];
    const float*     blv  = (const float*)d_in[7];

    const int hid  = in_sizes[3];             // 128
    const int outc = in_sizes[5];             // 64
    const int inc  = in_sizes[2] / hid;       // 256
    const int M    = in_sizes[0] / inc;       // 100000
    const int E    = in_sizes[1] / 2;         // 1.6M
    const long long* srcI = ei;
    const long long* dstI = ei + E;

    // carve workspace (~111 MB)
    char* ws = (char*)d_ws;
    size_t off = 0;
    auto carve = [&](size_t bytes) -> char* {
        char* p = ws + off;
        off = (off + bytes + 255) & ~(size_t)255;
        return p;
    };
    float*  deg  = (float*)carve((size_t)M * 4);
    float*  nrm  = (float*)carve((size_t)(E + M) * 4);
    __bf16* w1b  = (__bf16*)carve((size_t)inc * hid * 2);
    __bf16* wmub = (__bf16*)carve((size_t)hid * outc * 2);
    __bf16* wlvb = (__bf16*)carve((size_t)hid * outc * 2);
    float*  hpre = (float*)carve((size_t)M * hid * 4);   // reused for mu/lv pre
    float*  hagg = (float*)carve((size_t)M * hid * 4);
    float*  mupre = hpre;
    float*  lvpre = hpre + (size_t)M * outc;
    float*  out   = (float*)d_out;

    const int B = 256;
    auto G = [](long long n) { return (unsigned)((n + 255) / 256); };

    // init: zero accumulators, deg starts at 1 (self-loop pre-counted)
    fill_f32<<<G((long long)M * hid), B, 0, stream>>>(hagg, 0.0f, (long long)M * hid);
    fill_f32<<<G(2LL * M * outc), B, 0, stream>>>(out, 0.0f, 2LL * M * outc);
    fill_f32<<<G(M), B, 0, stream>>>(deg, 1.0f, M);

    // graph normalization
    degree_k<<<G(E), B, 0, stream>>>(dstI, deg, E);
    rsqrt_k<<<G(M), B, 0, stream>>>(deg, M);
    norm_k<<<G((long long)E + M), B, 0, stream>>>(srcI, dstI, deg, nrm, E, M);

    // pack weights into B-fragment-major bf16 (tiny, once)
    pack_w_k<<<G(inc * hid), B, 0, stream>>>(W1, w1b, inc, hid);
    pack_w_k<<<G(hid * outc), B, 0, stream>>>(Wmu, wmub, hid, outc);
    pack_w_k<<<G(hid * outc), B, 0, stream>>>(Wlv, wlvb, hid, outc);

    // layer 1: dense WMMA GEMM -> edge scatter -> bias + leaky relu
    gemm_x_w1<<<(M + 15) / 16, 256, 0, stream>>>(x, w1b, hpre, M);
    scatter_add<<<G((long long)(E + M) * (hid / 4)), B, 0, stream>>>(
        hpre, srcI, dstI, nrm, hagg, E, M, 5);
    finalize_h<<<G((long long)M * hid), B, 0, stream>>>(hagg, b1,
                                                        (long long)M * hid, hid);

    // layer 2: fused mu/logvar WMMA GEMM -> scatters into d_out -> bias
    gemm_h_w2<<<(M + 15) / 16, 256, 0, stream>>>(hagg, wmub, wlvb,
                                                 mupre, lvpre, M);
    scatter_add<<<G((long long)(E + M) * (outc / 4)), B, 0, stream>>>(
        mupre, srcI, dstI, nrm, out, E, M, 4);
    scatter_add<<<G((long long)(E + M) * (outc / 4)), B, 0, stream>>>(
        lvpre, srcI, dstI, nrm, out + (size_t)M * outc, E, M, 4);
    finalize_out<<<G(2LL * M * outc), B, 0, stream>>>(out, bmu, blv,
                                                      (long long)M, outc);
}
